// hyperBlock_30923764532124
// MI455X (gfx1250) — matrix operand order repaired and checked by reading, not compiled, not run on hardware
//
#include <hip/hip_runtime.h>
#include <hip/hip_bf16.h>
#include <math.h>

// ---------------- problem constants ----------------
#define DIMC   96
#define C4     384          // 4*DIMC
#define STYLE  128
#define LAT    8
#define KK     7
#define KV     343          // 7^3
#define PAD    3
#define BB     2
#define SPD    48
#define SP3    110592       // 48^3
#define EPS    1e-6f
#define HTS    129          // padded LDS row stride for the h tile (bank-conflict-free)

typedef __bf16 bf16_t;
typedef __attribute__((ext_vector_type(16))) __bf16 v16bf;
typedef __attribute__((ext_vector_type(8)))  __bf16 v8bf;
typedef __attribute__((ext_vector_type(8)))  float  v8f;

__device__ __forceinline__ v8f wmma_bf16(v16bf a, v16bf b, v8f c) {
    // D = A(16x32 bf16) x B(32x16 bf16) + C(16x16 f32)
    return __builtin_amdgcn_wmma_f32_16x16x32_bf16(
        /*neg_a=*/false, a, /*neg_b=*/false, b,
        /*c_mod=*/(short)0, c, /*reuse_a=*/false, /*reuse_b=*/false);
}

// Branch-free tanh-form GELU. Relative error ~1e-3, which is below the
// bf16 output rounding step (2^-8), so the stored result matches the
// exact-erf GELU at bf16 precision. Lowers to packed v_pk_* math +
// v_exp_f32 + v_rcp_f32 with no EXEC divergence.
__device__ __forceinline__ float gelu_fast(float v) {
    float u  = 0.7978845608028654f * v * (1.f + 0.044715f * v * v);
    float e  = __builtin_amdgcn_exp2f(2.885390081777927f * u);   // exp(2u)
    float th = 1.f - 2.f * __builtin_amdgcn_rcpf(e + 1.f);       // tanh(u)
    return 0.5f * v * (1.f + th);
}

// =====================================================================
// Kernel 1: z = s @ fcW^T + fcb   for the three hyper-nets.
// zbuf layout: [which(3)][B][LAT]
// =====================================================================
__global__ void compute_z_kernel(const float* __restrict__ s,
                                 const float* __restrict__ dw_fc_w,  const float* __restrict__ dw_fc_b,
                                 const float* __restrict__ pw1_fc_w, const float* __restrict__ pw1_fc_b,
                                 const float* __restrict__ pw2_fc_w, const float* __restrict__ pw2_fc_b,
                                 float* __restrict__ zbuf) {
    int t = threadIdx.x;
    if (t >= 3 * BB * LAT) return;
    int which = t / (BB * LAT);
    int r = t % (BB * LAT);
    int b = r / LAT, l = r % LAT;
    const float* fcw = (which == 0) ? dw_fc_w : (which == 1) ? pw1_fc_w : pw2_fc_w;
    const float* fcb = (which == 0) ? dw_fc_b : (which == 1) ? pw1_fc_b : pw2_fc_b;
    float acc = fcb[l];
    #pragma unroll 8
    for (int j = 0; j < STYLE; ++j) acc += s[b * STYLE + j] * fcw[l * STYLE + j];
    zbuf[t] = acc;
}

// =====================================================================
// Kernel 2: materialize per-batch weights from banks (dot over LAT=8).
//   w_dw (B,96,343) f32 ; b_dw (B,96) f32
//   w1   (B,384,96) bf16; b1   (B,384) f32
//   w2   (B,96,384) bf16; b2   (B,96) f32
// =====================================================================
#define T_WDW (BB*DIMC*KV)
#define T_BDW (BB*DIMC)
#define T_W1  (BB*C4*DIMC)
#define T_B1  (BB*C4)
#define T_W2  (BB*DIMC*C4)
#define T_B2  (BB*DIMC)
#define T_ALL (T_WDW+T_BDW+T_W1+T_B1+T_W2+T_B2)

__global__ void gen_weights_kernel(const float* __restrict__ dw_bank,  const float* __restrict__ dw_bias_bank,
                                   const float* __restrict__ pw1_bank, const float* __restrict__ pw1_bias_bank,
                                   const float* __restrict__ pw2_bank, const float* __restrict__ pw2_bias_bank,
                                   const float* __restrict__ zbuf,
                                   float* __restrict__ w_dw, float* __restrict__ b_dw,
                                   bf16_t* __restrict__ w1, float* __restrict__ b1,
                                   bf16_t* __restrict__ w2, float* __restrict__ b2) {
    int i = blockIdx.x * blockDim.x + threadIdx.x;
    if (i >= T_ALL) return;
    const float* z0 = zbuf;
    const float* z1 = zbuf + BB * LAT;
    const float* z2 = zbuf + 2 * BB * LAT;
    if (i < T_WDW) {
        int b = i / (DIMC * KV), r = i % (DIMC * KV);
        int o = r / KV, k = r % KV;
        float acc = 0.f;
        #pragma unroll
        for (int l = 0; l < LAT; ++l) acc += dw_bank[(o * KV + k) * LAT + l] * z0[b * LAT + l];
        w_dw[i] = acc;
        return;
    }
    i -= T_WDW;
    if (i < T_BDW) {
        int b = i / DIMC, o = i % DIMC;
        float acc = 0.f;
        #pragma unroll
        for (int l = 0; l < LAT; ++l) acc += dw_bias_bank[o * LAT + l] * z0[b * LAT + l];
        b_dw[i] = acc;
        return;
    }
    i -= T_BDW;
    if (i < T_W1) {
        int b = i / (C4 * DIMC), r = i % (C4 * DIMC);
        int o = r / DIMC, c = r % DIMC;
        float acc = 0.f;
        #pragma unroll
        for (int l = 0; l < LAT; ++l) acc += pw1_bank[(o * DIMC + c) * LAT + l] * z1[b * LAT + l];
        w1[i] = (bf16_t)acc;
        return;
    }
    i -= T_W1;
    if (i < T_B1) {
        int b = i / C4, o = i % C4;
        float acc = 0.f;
        #pragma unroll
        for (int l = 0; l < LAT; ++l) acc += pw1_bias_bank[o * LAT + l] * z1[b * LAT + l];
        b1[i] = acc;
        return;
    }
    i -= T_B1;
    if (i < T_W2) {
        int b = i / (DIMC * C4), r = i % (DIMC * C4);
        int o = r / C4, c = r % C4;
        float acc = 0.f;
        #pragma unroll
        for (int l = 0; l < LAT; ++l) acc += pw2_bank[(o * C4 + c) * LAT + l] * z2[b * LAT + l];
        w2[i] = (bf16_t)acc;
        return;
    }
    i -= T_W2;
    {
        int b = i / DIMC, o = i % DIMC;
        float acc = 0.f;
        #pragma unroll
        for (int l = 0; l < LAT; ++l) acc += pw2_bias_bank[o * LAT + l] * z2[b * LAT + l];
        b2[i] = acc;
    }
}

// =====================================================================
// Kernel 3: depthwise 7x7x7 conv, zero pad 3, + bias.
// One block = one (b,c) and one 8x8x8 output tile; 14^3 input tile +
// 343 weights in LDS. 256 threads, 2 outputs each, ADJACENT IN W:
// the two FMAs per tap hit consecutive LDS words -> v_pk_fma_f32 with
// broadcast weight, loads CSE across the kw unroll (8 loads / 14 uses),
// and the result pair stores as one b64.
// =====================================================================
__global__ __launch_bounds__(256) void dwconv_kernel(const float* __restrict__ x,
                                                     const float* __restrict__ w_dw,
                                                     const float* __restrict__ b_dw,
                                                     float* __restrict__ h) {
    __shared__ float tin[14 * 14 * 14];
    __shared__ float wsh[KV];
    __shared__ float bsh;
    const int tid = threadIdx.x;
    const int bc = blockIdx.y;            // b*96 + c
    const int tile = blockIdx.x;          // 0..215
    const int tw = tile % 6, th = (tile / 6) % 6, td = tile / 36;
    const size_t chanBase = (size_t)bc * SP3;

    // load padded input tile (zero OOB)
    for (int i = tid; i < 14 * 14 * 14; i += 256) {
        int wz = i % 14, hy = (i / 14) % 14, dz = i / 196;
        int gw = tw * 8 + wz - PAD, gh = th * 8 + hy - PAD, gd = td * 8 + dz - PAD;
        float v = 0.f;
        if ((unsigned)gw < SPD && (unsigned)gh < SPD && (unsigned)gd < SPD)
            v = x[chanBase + ((size_t)gd * SPD + gh) * SPD + gw];
        tin[i] = v;
    }
    for (int i = tid; i < KV; i += 256) wsh[i] = w_dw[(size_t)bc * KV + i];
    if (tid == 0) bsh = b_dw[bc];
    __syncthreads();

    // two outputs per thread, adjacent in w: (od,oh,2p) and (od,oh,2p+1)
    const int p  = tid & 3;               // w pair index: w = 2p, 2p+1
    const int oh = (tid >> 2) & 7;
    const int od = tid >> 5;              // 0..7
    const int ow = p * 2;
    float acc0 = bsh, acc1 = bsh;
    #pragma unroll 1
    for (int kd = 0; kd < KK; ++kd) {
        #pragma unroll 1
        for (int kh = 0; kh < KK; ++kh) {
            const int base = (od + kd) * 196 + (oh + kh) * 14 + ow;
            const float* wp = &wsh[(kd * KK + kh) * KK];
            #pragma unroll
            for (int kw = 0; kw < KK; ++kw) {
                float w = wp[kw];
                acc0 += w * tin[base + kw];
                acc1 += w * tin[base + kw + 1];
            }
        }
    }
    float2 st; st.x = acc0; st.y = acc1;
    const int gw = tw * 8 + ow, gh = th * 8 + oh, gd = td * 8 + od;
    *(float2*)&h[chanBase + ((size_t)gd * SPD + gh) * SPD + gw] = st;
}

// =====================================================================
// Kernel 4: FUSED  channel-LayerNorm + pw1 GEMM + fast GELU -> bf16 h2.
// h2 uses an n-major internal layout  h2[b][n][o]  (o = 0..383 contiguous)
// so the epilogue writes one 16B global_store_b128 per M tile and pw2's
// B-fragments are contiguous 32B loads.
// =====================================================================
__global__ __launch_bounds__(256) void pw1_fused_kernel(int b,
                                                        const float* __restrict__ h,
                                                        const float* __restrict__ ln_w,
                                                        const float* __restrict__ ln_b,
                                                        const bf16_t* __restrict__ w1,
                                                        const float* __restrict__ b1,
                                                        bf16_t* __restrict__ h2) {
    __shared__ bf16_t lw[C4 * DIMC];        // 72 KB
    __shared__ float  ht[DIMC * HTS];       // 96x129 f32 = 49.5 KB
    const int tid = threadIdx.x;
    const int nbase = blockIdx.x * 128;

    // stage W1 for this batch
    {
        const uint32_t* src = (const uint32_t*)(w1 + (size_t)b * C4 * DIMC);
        uint32_t* dst = (uint32_t*)lw;
        #pragma unroll 4
        for (int i = tid; i < C4 * DIMC / 2; i += 256) dst[i] = src[i];
    }
    // stage h slab (coalesced rows of 128 f32)
    {
        const float* hs = h + (size_t)b * DIMC * SP3 + nbase;
        for (int i = tid; i < DIMC * 128; i += 256) {
            int c = i >> 7, n = i & 127;
            ht[c * HTS + n] = hs[(size_t)c * SP3 + n];
        }
    }
    __syncthreads();

    // in-LDS LayerNorm: one thread per voxel column
    if (tid < 128) {
        const int n = tid;
        float sum = 0.f, sq = 0.f;
        #pragma unroll 8
        for (int c = 0; c < DIMC; ++c) {
            float v = ht[c * HTS + n];
            sum += v; sq += v * v;
        }
        const float mean = sum * (1.f / DIMC);
        const float var  = sq * (1.f / DIMC) - mean * mean;
        const float inv  = 1.f / sqrtf(var + EPS);
        #pragma unroll 8
        for (int c = 0; c < DIMC; ++c) {
            float v = ht[c * HTS + n];
            ht[c * HTS + n] = (v - mean) * inv * ln_w[c] + ln_b[c];
        }
    }
    __syncthreads();

    const int wave = tid >> 5, lane = tid & 31;
    const int col = lane & 15;
    const int hi  = lane >> 4;
    const int kbB = hi * 16;                 // B frag: K = kbB + j
    const int row = lane & 15;               // A frag row
    const int kbA = hi * 16;                 // A frag: K = kbA + j
    const int rowHi = hi * 8;                // C/D: M = r + rowHi
    const int ncol = wave * 16 + col;        // column within the 128-voxel tile
    const int nglob = nbase + ncol;

    // B fragments for K = 0..95 from normalized LDS tile (f32 -> bf16)
    v16bf bf0, bf1, bf2;
    #pragma unroll
    for (int j = 0; j < 16; ++j) {
        bf0[j] = (bf16_t)ht[(kbB + j)      * HTS + ncol];
        bf1[j] = (bf16_t)ht[(32 + kbB + j) * HTS + ncol];
        bf2[j] = (bf16_t)ht[(64 + kbB + j) * HTS + ncol];
    }

    // n-major output row for this lane: 384 contiguous bf16
    bf16_t* hrow = h2 + ((size_t)b * SP3 + nglob) * C4 + rowHi;
    const float* b1p = b1 + b * C4 + rowHi;

    #pragma unroll 1
    for (int m = 0; m < C4 / 16; ++m) {      // 24 M tiles
        v8f acc = {0.f, 0.f, 0.f, 0.f, 0.f, 0.f, 0.f, 0.f};
        const bf16_t* par = &lw[(m * 16 + row) * DIMC];
        v16bf a;
        #pragma unroll
        for (int kc = 0; kc < 3; ++kc) {
            const bf16_t* pa = par + kc * 32 + kbA;
            #pragma unroll
            for (int j = 0; j < 8; ++j) { a[j] = pa[j]; a[8 + j] = pa[8 + j]; }
            acc = wmma_bf16(a, (kc == 0) ? bf0 : (kc == 1) ? bf1 : bf2, acc);
        }
        v8bf st;
        #pragma unroll
        for (int r = 0; r < 8; ++r) {
            float v = acc[r] + b1p[m * 16 + r];
            st[r] = (bf16_t)gelu_fast(v);
        }
        *(v8bf*)(hrow + m * 16) = st;        // one 16B store per M tile
    }
}

// =====================================================================
// Kernel 5: pw2 GEMM (96x384 @ 384xN) + bias, out = x + gamma*h.
// h2 is n-major [n][384]: each lane's B-fragment is 32 contiguous bytes
// (one v16bf load -> global_load_b128 pair). 6 f32 accumulators across
// 12 K chunks; W2 staged in LDS; `out` streamed with NT stores.
// =====================================================================
__global__ __launch_bounds__(256) void pw2_out_kernel(int b,
                                                      const bf16_t* __restrict__ h2,
                                                      const bf16_t* __restrict__ w2,
                                                      const float* __restrict__ b2,
                                                      const float* __restrict__ x,
                                                      const float* __restrict__ gamma,
                                                      float* __restrict__ out) {
    __shared__ bf16_t lw[DIMC * C4];          // 96x384 bf16 = 72 KB
    const int tid = threadIdx.x;
    {
        const uint32_t* src = (const uint32_t*)(w2 + (size_t)b * DIMC * C4);
        uint32_t* dst = (uint32_t*)lw;
        #pragma unroll 4
        for (int i = tid; i < DIMC * C4 / 2; i += 256) dst[i] = src[i];
    }
    __syncthreads();

    const int wave = tid >> 5, lane = tid & 31;
    const int n0 = (blockIdx.x * 8 + wave) * 16;
    const int col = lane & 15;
    const int hi  = lane >> 4;
    const int kbB = hi * 16;
    const int row = lane & 15;
    const int kbA = hi * 16;
    const int rowHi = hi * 8;

    // lane's h2 column: 384 contiguous bf16 starting at kbB
    const bf16_t* hcol = h2 + ((size_t)b * SP3 + n0 + col) * C4 + kbB;

    v8f acc0 = {0,0,0,0,0,0,0,0}, acc1 = acc0, acc2 = acc0, acc3 = acc0, acc4 = acc0, acc5 = acc0;

    #pragma unroll 1
    for (int kc = 0; kc < C4 / 32; ++kc) {    // 12 K chunks
        v16bf bfr = *(const v16bf*)(hcol + kc * 32);   // 32B contiguous
        #pragma unroll
        for (int m = 0; m < DIMC / 16; ++m) { // 6 M tiles
            const bf16_t* pa = &lw[(m * 16 + row) * C4 + kc * 32 + kbA];
            v16bf a;
            #pragma unroll
            for (int j = 0; j < 8; ++j) { a[j] = pa[j]; a[8 + j] = pa[8 + j]; }
            v8f& acc = (m == 0) ? acc0 : (m == 1) ? acc1 : (m == 2) ? acc2
                      : (m == 3) ? acc3 : (m == 4) ? acc4 : acc5;
            acc = wmma_bf16(a, bfr, acc);
        }
    }
    #pragma unroll
    for (int m = 0; m < DIMC / 16; ++m) {
        const v8f& acc = (m == 0) ? acc0 : (m == 1) ? acc1 : (m == 2) ? acc2
                        : (m == 3) ? acc3 : (m == 4) ? acc4 : acc5;
        #pragma unroll
        for (int r = 0; r < 8; ++r) {
            const int o = m * 16 + r + rowHi;
            const size_t xi = ((size_t)b * DIMC + o) * SP3 + n0 + col;
            float v = acc[r] + b2[b * DIMC + o];
            __builtin_nontemporal_store(x[xi] + gamma[o] * v, &out[xi]);
        }
    }
}

// =====================================================================
// Host launcher
// =====================================================================
static inline size_t align256(size_t v) { return (v + 255) & ~(size_t)255; }

extern "C" void kernel_launch(void* const* d_in, const int* in_sizes, int n_in,
                              void* d_out, int out_size, void* d_ws, size_t ws_size,
                              hipStream_t stream) {
    (void)in_sizes; (void)n_in; (void)out_size; (void)ws_size;
    const float* x            = (const float*)d_in[0];
    const float* s            = (const float*)d_in[1];
    const float* ln_w         = (const float*)d_in[2];
    const float* ln_b         = (const float*)d_in[3];
    const float* gamma        = (const float*)d_in[4];
    const float* dw_fc_w      = (const float*)d_in[5];
    const float* dw_fc_b      = (const float*)d_in[6];
    const float* dw_bank      = (const float*)d_in[7];
    const float* dw_bias_bank = (const float*)d_in[8];
    const float* pw1_fc_w     = (const float*)d_in[9];
    const float* pw1_fc_b     = (const float*)d_in[10];
    const float* pw1_bank     = (const float*)d_in[11];
    const float* pw1_bias_bank= (const float*)d_in[12];
    const float* pw2_fc_w     = (const float*)d_in[13];
    const float* pw2_fc_b     = (const float*)d_in[14];
    const float* pw2_bank     = (const float*)d_in[15];
    const float* pw2_bias_bank= (const float*)d_in[16];
    float* out = (float*)d_out;

    // workspace layout (h and h2 are live simultaneously inside pw1 -> disjoint)
    char* ws = (char*)d_ws;
    size_t off = 0;
    float*  zbuf = (float*)(ws + off);  off = align256(off + 3 * BB * LAT * sizeof(float));
    float*  w_dw = (float*)(ws + off);  off = align256(off + (size_t)T_WDW * sizeof(float));
    float*  b_dw = (float*)(ws + off);  off = align256(off + (size_t)T_BDW * sizeof(float));
    float*  b1   = (float*)(ws + off);  off = align256(off + (size_t)T_B1  * sizeof(float));
    float*  b2   = (float*)(ws + off);  off = align256(off + (size_t)T_B2  * sizeof(float));
    bf16_t* w1   = (bf16_t*)(ws + off); off = align256(off + (size_t)T_W1  * sizeof(bf16_t));
    bf16_t* w2   = (bf16_t*)(ws + off); off = align256(off + (size_t)T_W2  * sizeof(bf16_t));
    float*  h    = (float*)(ws + off);  off = align256(off + (size_t)BB * DIMC * SP3 * sizeof(float));
    bf16_t* h2   = (bf16_t*)(ws + off); off = align256(off + (size_t)BB * C4 * SP3 * sizeof(bf16_t));

    // 1) hyper-net latents
    compute_z_kernel<<<1, 64, 0, stream>>>(s, dw_fc_w, dw_fc_b, pw1_fc_w, pw1_fc_b,
                                           pw2_fc_w, pw2_fc_b, zbuf);
    // 2) materialize per-batch weights
    gen_weights_kernel<<<(T_ALL + 255) / 256, 256, 0, stream>>>(
        dw_bank, dw_bias_bank, pw1_bank, pw1_bias_bank, pw2_bank, pw2_bias_bank,
        zbuf, w_dw, b_dw, w1, b1, w2, b2);
    // 3) depthwise 7^3 conv + bias (both batches)
    dwconv_kernel<<<dim3(216, BB * DIMC), 256, 0, stream>>>(x, w_dw, b_dw, h);
    // 4+5) per-batch: fused LN+pw1+GELU (WMMA), then pw2+residual (WMMA)
    //      -> h2[b] (85 MB) stays L2-resident between the two GEMMs.
    for (int b = 0; b < BB; ++b) {
        pw1_fused_kernel<<<dim3(SP3 / 128), 256, 0, stream>>>(b, h, ln_w, ln_b, w1, b1, h2);
        pw2_out_kernel<<<dim3(SP3 / 128), 256, 0, stream>>>(b, h2, w2, b2, x, gamma, out);
    }
}
